// NanoPET_37847251812815
// MI455X (gfx1250) — compile-verified, matrix operand order checked by reading, not run on hardware
//
#include <hip/hip_runtime.h>

// ----------------------------- constants --------------------------------
#define N_NODES 4096
#define MAX_E   64
#define DIM     128
#define NHEAD   4
#define DHEAD   32
#define NLAYER  2
#define N_EDGES 131072

typedef __bf16 bf16;
typedef __attribute__((ext_vector_type(16))) __bf16 v16bf;
typedef __attribute__((ext_vector_type(8)))  float  v8f;

// bf16-transposed weight buffer layout (element offsets)
constexpr size_t W_QKV = 0;                        // [384][128]
constexpr size_t W_OUT = W_QKV + 128 * 384;        // [128][128]
constexpr size_t W_M1  = W_OUT + 128 * 128;        // [512][128]
constexpr size_t W_M2  = W_M1  + 128 * 512;        // [128][512]
constexpr size_t PL    = W_M2  + 512 * 128;        // 196608 elems / layer
constexpr size_t PT    = 2 * PL;                   // per transformer
constexpr size_t W_GNN = 4 * PL;                   // [128][256]
constexpr size_t WBUF_ELEMS = W_GNN + 128 * 256;

// workspace layout (byte offsets)
constexpr size_t OFF_RM   = 0;
constexpr size_t OFF_FEAT = OFF_RM   + (size_t)N_NODES * MAX_E * 4;
constexpr size_t OFF_T1IN = OFF_FEAT + (size_t)N_NODES * MAX_E * DIM * 4;
constexpr size_t OFF_WBUF = OFF_T1IN + (size_t)N_NODES * MAX_E * DIM * 4;
constexpr size_t OFF_ENCW = OFF_WBUF + WBUF_ELEMS * 2;

// ------------------------ async global<->LDS copy -----------------------
// CDNA5 GLOBAL_LOAD_ASYNC_TO_LDS_B128 / GLOBAL_STORE_ASYNC_FROM_LDS_B128
// (tracked with ASYNCcnt). The generic address of a __shared__ object has
// the wave-relative LDS byte offset in its low 32 bits (ISA 10.2 aperture
// mapping), which is exactly what the VDST/VSRC LDS-address operand wants.
__device__ __forceinline__ void async_load_b128(const void* gsrc, void* ldst) {
    unsigned int laddr = (unsigned int)(unsigned long long)ldst;
    unsigned long long gaddr = (unsigned long long)gsrc;
    asm volatile("global_load_async_to_lds_b128 %0, %1, off"
                 :: "v"(laddr), "v"(gaddr)
                 : "memory");
}
__device__ __forceinline__ void async_store_b128(void* gdst, const void* lsrc) {
    unsigned int laddr = (unsigned int)(unsigned long long)lsrc;
    unsigned long long gaddr = (unsigned long long)gdst;
    asm volatile("global_store_async_from_lds_b128 %0, %1, off"
                 :: "v"(gaddr), "v"(laddr)
                 : "memory");
}
__device__ __forceinline__ void async_wait_all() {
    asm volatile("s_wait_asynccnt 0x0" ::: "memory");
}

// ----------------------------- WMMA helpers -----------------------------
__device__ __forceinline__ v8f vzero8() {
    v8f z = {0.f, 0.f, 0.f, 0.f, 0.f, 0.f, 0.f, 0.f};
    return z;
}

__device__ __forceinline__ v8f wmma_bf(v16bf a, v16bf b, v8f c) {
    // D(16x16,f32) = A(16x32,bf16) * B(32x16,bf16) + C
    return __builtin_amdgcn_wmma_f32_16x16x32_bf16(
        /*neg_a=*/false, a, /*neg_b=*/false, b,
        /*c_mod=*/(short)0, c, /*reuse_a=*/false, /*reuse_b=*/false);
}

// 16x32 bf16 fragment, row-major source with leading dim `ld`.
// Lane m (0-15) holds row row0+m, K subsets per ISA 16-bit A/B layout:
// lanes 0-15: K = k0+0..7 and k0+16..23 ; lanes 16-31: K = k0+8..15 and k0+24..31.
__device__ __forceinline__ v16bf frag16(const bf16* base, int ld, int row0,
                                        int k0, int lane) {
    const bf16* p = base + (size_t)(row0 + (lane & 15)) * ld + k0 + ((lane >> 4) << 3);
    v16bf r;
    reinterpret_cast<uint4*>(&r)[0] = *reinterpret_cast<const uint4*>(p);
    reinterpret_cast<uint4*>(&r)[1] = *reinterpret_cast<const uint4*>(p + 16);
    return r;
}

// C(16x16) += A(rows m0..m0+15 of lda-strided bf16) x Bt(rows = output cols, ldb-strided)
__device__ __forceinline__ v8f gemm_tile(const bf16* A, int lda, int m0,
                                         const bf16* Bt, int ldb, int K, int lane) {
    v8f acc = vzero8();
    for (int k0 = 0; k0 < K; k0 += 32)
        acc = wmma_bf(frag16(A, lda, m0, k0, lane),
                      frag16(Bt, ldb, 0, k0, lane), acc);
    return acc;
}

__device__ __forceinline__ void layer_norm_row(const float* x, bf16* o,
                                               const float* wv, const float* bv) {
    float m = 0.f;
    for (int j = 0; j < DIM; ++j) m += x[j];
    m *= (1.0f / DIM);
    float v = 0.f;
    for (int j = 0; j < DIM; ++j) { float d = x[j] - m; v += d * d; }
    v *= (1.0f / DIM);
    float rs = rsqrtf(v + 1e-5f);
    for (int j = 0; j < DIM; ++j)
        o[j] = (bf16)((x[j] - m) * rs * wv[j] + bv[j]);
}

// ------------------- weight prep kernels --------------------------------
__global__ void transposew_kernel(const float* __restrict__ src, bf16* __restrict__ dst,
                                  int K, int N) {
    int idx = blockIdx.x * 256 + threadIdx.x;
    if (idx >= K * N) return;
    int n = idx / K, k = idx - n * K;
    dst[idx] = (bf16)src[k * N + n];     // dst[n][k] = src[k][n]
}

// Fold enc(cart/center/neighbor concat) @ enc_comp_w into small fused tables.
__global__ void fuse_enc_kernel(const float* __restrict__ cart_w,
                                const float* __restrict__ ce,
                                const float* __restrict__ ne,
                                const float* __restrict__ comp,
                                float* __restrict__ Af, float* __restrict__ cef,
                                float* __restrict__ nefw) {
    int j = threadIdx.x;  // 128
    for (int i = 0; i < 3; ++i) {
        float s = 0.f;
        for (int k = 0; k < 128; ++k) s += cart_w[i * 128 + k] * comp[k * 128 + j];
        Af[i * 128 + j] = s;
    }
    for (int sp = 0; sp < 4; ++sp) {
        float s1 = 0.f, s2 = 0.f;
        for (int k = 0; k < 128; ++k) {
            s1 += ce[sp * 128 + k] * comp[(128 + k) * 128 + j];
            s2 += ne[sp * 128 + k] * comp[(256 + k) * 128 + j];
        }
        cef[sp * 128 + j]  = s1;
        nefw[sp * 128 + j] = s2;
    }
}

// ------------------------------ encoder ---------------------------------
__global__ void encoder_kernel(const float* __restrict__ ev,
                               const int* __restrict__ nef_idx,
                               const int* __restrict__ nef_mask,
                               const int* __restrict__ centers,
                               const int* __restrict__ neighbors,
                               const int* __restrict__ species,
                               const float* __restrict__ Af,
                               const float* __restrict__ cef,
                               const float* __restrict__ nefw,
                               float* __restrict__ feat, float* __restrict__ rm) {
    int n = blockIdx.x >> 6, s = blockIdx.x & 63, j = threadIdx.x;  // 128 thr
    int slot = (n << 6) + s;
    int e = nef_idx[slot];
    float x0 = ev[e * 3 + 0], x1 = ev[e * 3 + 1], x2 = ev[e * 3 + 2];
    if (j == 0) {
        float r = sqrtf(x0 * x0 + x1 * x1 + x2 * x2);
        float rad = (r < 3.0f)
                        ? 1.0f
                        : 0.5f * (cosf(3.14159265358979f * (r - 3.0f) * 0.5f) + 1.0f);
        if (r >= 5.0f) rad = 0.0f;
        rm[slot] = nef_mask[slot] ? rad : 0.0f;
    }
    int ic = species[centers[e]], in_ = species[neighbors[e]];
    feat[(size_t)slot * DIM + j] = x0 * Af[j] + x1 * Af[128 + j] + x2 * Af[256 + j] +
                                   cef[ic * 128 + j] + nefw[in_ * 128 + j];
}

// --------------------- per-node transformer (L=2) -----------------------
__global__ __launch_bounds__(256) void transformer_kernel(
    float* __restrict__ xg, const float* __restrict__ rmg,
    const bf16* __restrict__ wT,    // this transformer's bf16-transposed weights
    const float* __restrict__ ln1w, const float* __restrict__ ln1b,
    const float* __restrict__ qkvb, const float* __restrict__ outbv,
    const float* __restrict__ ln2w, const float* __restrict__ ln2b,
    const float* __restrict__ m1b,  const float* __restrict__ m2b) {
    __shared__ float xf[MAX_E][DIM];     // residual stream (fp32)
    __shared__ bf16  aOp[MAX_E][DIM];    // LN output / GEMM A operand
    __shared__ bf16  qk[MAX_E][256];     // q,k rows (also MLP hidden chunk)
    __shared__ bf16  vT[DIM][MAX_E];     // v transposed (dh-major)
    __shared__ float sc[MAX_E][MAX_E];   // per-head scores
    __shared__ bf16  pb[MAX_E][MAX_E];   // attention probabilities
    __shared__ bf16  ob[MAX_E][DIM];     // attention output
    __shared__ float rms[MAX_E];

    const int tid = threadIdx.x, lane = tid & 31, w = tid >> 5;
    const int node = blockIdx.x;
    float* xrow = xg + (size_t)node * MAX_E * DIM;

    if (tid == 0) {
        __builtin_prefetch(wT, 0, 3);
        __builtin_prefetch(wT + PL, 0, 3);
    }
    // async DMA the 32KB residual tile into LDS (8 x B128 per thread)
    for (int i = tid; i < (MAX_E * DIM) / 4; i += 256)
        async_load_b128(xrow + (size_t)i * 4, &xf[0][0] + (size_t)i * 4);
    if (tid < MAX_E) rms[tid] = rmg[(size_t)node * MAX_E + tid];
    async_wait_all();
    __syncthreads();

    for (int l = 0; l < NLAYER; ++l) {
        const bf16* qkvT = wT + (size_t)l * PL + W_QKV;
        const bf16* outT = wT + (size_t)l * PL + W_OUT;
        const bf16* m1T  = wT + (size_t)l * PL + W_M1;
        const bf16* m2T  = wT + (size_t)l * PL + W_M2;
        const float* l1w = ln1w + l * DIM, * l1b = ln1b + l * DIM;
        const float* l2w = ln2w + l * DIM, * l2b = ln2b + l * DIM;
        const float* qb  = qkvb + l * 384, * obias = outbv + l * DIM;
        const float* b1  = m1b + l * 512,  * b2 = m2b + l * DIM;

        // ---- LN1 ----
        if (tid < MAX_E) layer_norm_row(xf[tid], aOp[tid], l1w, l1b);
        __syncthreads();

        // ---- QKV q/k part: (64x128)@(128x256), 64 tiles, 8 per wave ----
        for (int i = 0; i < 8; ++i) {
            int t = w + (i << 3);
            int m0 = (t >> 4) << 4, n0 = (t & 15) << 4;
            v8f acc = gemm_tile(&aOp[0][0], DIM, m0, qkvT + (size_t)n0 * DIM, DIM, DIM, lane);
            int col = n0 + (lane & 15);
            float bias = qb[col];
            for (int r = 0; r < 8; ++r) {
                int m = m0 + r + ((lane >> 4) << 3);
                qk[m][col] = (bf16)(acc[r] + bias);   // q,k row-major
            }
        }
        // ---- QKV v part: (64x128)@(128x128), 32 tiles, 4 per wave; store v^T
        for (int i = 0; i < 4; ++i) {
            int t = w + (i << 3);
            int m0 = (t >> 3) << 4, n0v = (t & 7) << 4;
            v8f acc = gemm_tile(&aOp[0][0], DIM, m0,
                                qkvT + (size_t)(256 + n0v) * DIM, DIM, DIM, lane);
            int colv = n0v + (lane & 15);
            float bias = qb[256 + colv];
            for (int r = 0; r < 8; ++r) {
                int m = m0 + r + ((lane >> 4) << 3);
                vT[colv][m] = (bf16)(acc[r] + bias);  // v transposed
            }
        }
        __syncthreads();

        // ---- attention per head ----
        const float iscale = 0.17677669529663687f;  // 1/sqrt(32)
        for (int h = 0; h < NHEAD; ++h) {
            // scores (64x64): 16 tiles, 2 per wave; single K=32 WMMA each
            for (int i = 0; i < 2; ++i) {
                int t = w + (i << 3);
                int m0 = (t >> 2) << 4, n0 = (t & 3) << 4;
                v8f acc = wmma_bf(frag16(&qk[0][0] + h * DHEAD, 256, m0, 0, lane),
                                  frag16(&qk[0][0] + 128 + h * DHEAD, 256, n0, 0, lane),
                                  vzero8());
                int n = n0 + (lane & 15);
                for (int r = 0; r < 8; ++r)
                    sc[m0 + r + ((lane >> 4) << 3)][n] = acc[r] * iscale;
            }
            __syncthreads();
            // masked softmax: a = softmax(s); a *= rm; a /= (sum a + 1e-9)
            if (tid < MAX_E) {
                float mx = -1e30f;
                for (int n2 = 0; n2 < MAX_E; ++n2) mx = fmaxf(mx, sc[tid][n2]);
                float se = 0.f, sw = 0.f;
                for (int n2 = 0; n2 < MAX_E; ++n2) {
                    float e = expf(sc[tid][n2] - mx);
                    sc[tid][n2] = e;
                    se += e;
                    sw += e * rms[n2];
                }
                float inv = 1.0f / (sw + 1e-9f * se);
                for (int n2 = 0; n2 < MAX_E; ++n2)
                    pb[tid][n2] = (bf16)(sc[tid][n2] * rms[n2] * inv);
            }
            __syncthreads();
            // o_h = probs(64x64) @ v(64x32): 8 tiles, 1 per wave, K=64
            {
                int m0 = (w >> 1) << 4, n0 = (w & 1) << 4;
                v8f acc = gemm_tile(&pb[0][0], MAX_E, m0, &vT[h * DHEAD + n0][0],
                                    MAX_E, MAX_E, lane);
                int n = h * DHEAD + n0 + (lane & 15);
                for (int r = 0; r < 8; ++r)
                    ob[m0 + r + ((lane >> 4) << 3)][n] = (bf16)acc[r];
            }
            __syncthreads();
        }

        // ---- out-proj: x += o @ out_w + b : 32 tiles, 4 per wave ----
        for (int i = 0; i < 4; ++i) {
            int t = w + (i << 3);
            int m0 = (t >> 3) << 4, n0 = (t & 7) << 4;
            v8f acc = gemm_tile(&ob[0][0], DIM, m0, outT + (size_t)n0 * DIM, DIM, DIM, lane);
            int n = n0 + (lane & 15);
            float bias = obias[n];
            for (int r = 0; r < 8; ++r)
                xf[m0 + r + ((lane >> 4) << 3)][n] += acc[r] + bias;
        }
        __syncthreads();

        // ---- LN2 ----
        if (tid < MAX_E) layer_norm_row(xf[tid], aOp[tid], l2w, l2b);
        __syncthreads();

        // ---- MLP: hidden 512 in two 256-chunks; second GEMM accumulated in VGPRs
        v8f acc2[4];
        for (int i = 0; i < 4; ++i) acc2[i] = vzero8();
        for (int c = 0; c < 2; ++c) {
            // hidden chunk (64x256): 64 tiles, 8 per wave
            for (int i = 0; i < 8; ++i) {
                int t = w + (i << 3);
                int m0 = (t >> 4) << 4, n0 = (t & 15) << 4;
                v8f acc = gemm_tile(&aOp[0][0], DIM, m0,
                                    m1T + (size_t)(c * 256 + n0) * DIM, DIM, DIM, lane);
                int nloc = n0 + (lane & 15);
                float bias = b1[c * 256 + nloc];
                for (int r = 0; r < 8; ++r) {
                    float v = acc[r] + bias;
                    float g = 0.5f * v *
                              (1.0f + tanhf(0.7978845608f * (v + 0.044715f * v * v * v)));
                    qk[m0 + r + ((lane >> 4) << 3)][nloc] = (bf16)g;  // reuse qk as hidden
                }
            }
            __syncthreads();
            // partial second GEMM: K = 256 of this chunk
            for (int i = 0; i < 4; ++i) {
                int t = w + (i << 3);
                int m0 = (t >> 3) << 4, n0 = (t & 7) << 4;
                for (int k0 = 0; k0 < 256; k0 += 32)
                    acc2[i] = wmma_bf(frag16(&qk[0][0], 256, m0, k0, lane),
                                      frag16(m2T + (size_t)n0 * 512 + c * 256, 512, 0, k0, lane),
                                      acc2[i]);
            }
            __syncthreads();
        }
        for (int i = 0; i < 4; ++i) {
            int t = w + (i << 3);
            int m0 = (t >> 3) << 4, n0 = (t & 7) << 4;
            int n = n0 + (lane & 15);
            float bias = b2[n];
            for (int r = 0; r < 8; ++r)
                xf[m0 + r + ((lane >> 4) << 3)][n] += acc2[i][r] + bias;
        }
        __syncthreads();
    }

    // async DMA the residual tile back LDS->global (8 x B128 per thread)
    for (int i = tid; i < (MAX_E * DIM) / 4; i += 256)
        async_store_b128(xrow + (size_t)i * 4, &xf[0][0] + (size_t)i * 4);
    async_wait_all();
}

// ------------- GNN message passing: concat-gather GEMM + NEF scatter ----
__global__ __launch_bounds__(256) void gnn_kernel(const float* __restrict__ feat,
                                                  const int* __restrict__ centers,
                                                  const int* __restrict__ n2e,
                                                  const int* __restrict__ corr,
                                                  const bf16* __restrict__ gnnT,
                                                  float* __restrict__ t1in) {
    __shared__ int  rowA[64], rowB[64];
    __shared__ bf16 Ab[64][256];
    int tid = threadIdx.x, lane = tid & 31, w = tid >> 5;
    int e0 = blockIdx.x << 6;
    if (tid < 64) {
        int e = e0 + tid;
        rowA[tid] = centers[e] * 64 + n2e[e];      // NEF slot of edge e
        int ce = corr[e];
        rowB[tid] = centers[ce] * 64 + n2e[ce];    // slot of reverse edge
    }
    __syncthreads();
    // stage A = concat(feat[e], feat[corr(e)]) as bf16
    for (int idx = tid; idx < 64 * 256; idx += 256) {
        int r = idx >> 8, c = idx & 255;
        float v = (c < 128) ? feat[(size_t)rowA[r] * DIM + c]
                            : feat[(size_t)rowB[r] * DIM + (c - 128)];
        Ab[r][c] = (bf16)v;
    }
    __syncthreads();
    // (64x256)@(256x128): 32 tiles, 4 per wave, K=256; scatter back to NEF
    for (int i = 0; i < 4; ++i) {
        int t = w + (i << 3);
        int m0 = (t >> 3) << 4, n0 = (t & 7) << 4;
        v8f acc = gemm_tile(&Ab[0][0], 256, m0, gnnT + (size_t)n0 * 256, 256, 256, lane);
        int n = n0 + (lane & 15);
        for (int r = 0; r < 8; ++r) {
            int m = m0 + r + ((lane >> 4) << 3);
            t1in[(size_t)rowA[m] * DIM + n] = acc[r];
        }
    }
}

// ------------------------------ readout ---------------------------------
__global__ void readout_kernel(const float* __restrict__ feat,
                               const float* __restrict__ t1in,
                               const float* __restrict__ rm,
                               const float* __restrict__ last_w,
                               const float* __restrict__ comp_w,
                               const int* __restrict__ species,
                               float* __restrict__ out) {
    __shared__ float red[128];
    int nd = blockIdx.x, j = threadIdx.x;  // 128 threads
    const float* f = feat + (size_t)nd * MAX_E * DIM;
    const float* g = t1in + (size_t)nd * MAX_E * DIM;
    const float* rr = rm + (size_t)nd * MAX_E;
    float acc = 0.f;
    for (int s = 0; s < MAX_E; ++s)
        acc += (f[s * DIM + j] + g[s * DIM + j]) * rr[s];
    red[j] = acc * last_w[j];
    __syncthreads();
    for (int st = 64; st > 0; st >>= 1) {
        if (j < st) red[j] += red[j + st];
        __syncthreads();
    }
    if (j == 0) out[nd] = red[0] + comp_w[species[nd]];
}

// ------------------------------ launcher --------------------------------
extern "C" void kernel_launch(void* const* d_in, const int* in_sizes, int n_in,
                              void* d_out, int out_size, void* d_ws, size_t ws_size,
                              hipStream_t stream) {
    (void)in_sizes; (void)n_in; (void)out_size; (void)ws_size;

    const float* edge_vectors = (const float*)d_in[0];
    const float* last_w       = (const float*)d_in[6];
    const float* comp_w       = (const float*)d_in[7];
    const int*   species      = (const int*)d_in[8];
    const int*   centers      = (const int*)d_in[9];
    const int*   neighbors    = (const int*)d_in[10];
    const int*   nef_indices  = (const int*)d_in[11];
    const int*   nef_mask     = (const int*)d_in[12];
    const int*   n2e          = (const int*)d_in[13];
    const int*   corr         = (const int*)d_in[14];

    char*  ws    = (char*)d_ws;
    float* rm    = (float*)(ws + OFF_RM);
    float* feat  = (float*)(ws + OFF_FEAT);
    float* t1in  = (float*)(ws + OFF_T1IN);
    bf16*  wbuf  = (bf16*)(ws + OFF_WBUF);
    float* Af    = (float*)(ws + OFF_ENCW);
    float* cef   = Af + 3 * 128;
    float* nefw  = cef + 4 * 128;

    // 1) convert weights to bf16, transposed (N x K)
    struct WD { const float* src; size_t off; int K, N; };
    WD wds[17];
    int nw = 0;
    for (int t = 0; t < 2; ++t) {
        int b = 15 + t * 12;
        const float* qkvw = (const float*)d_in[b + 2];
        const float* outw = (const float*)d_in[b + 4];
        const float* m1w  = (const float*)d_in[b + 8];
        const float* m2w  = (const float*)d_in[b + 10];
        for (int l = 0; l < 2; ++l) {
            size_t base = (size_t)t * PT + (size_t)l * PL;
            wds[nw++] = { qkvw + (size_t)l * 128 * 384, base + W_QKV, 128, 384 };
            wds[nw++] = { outw + (size_t)l * 128 * 128, base + W_OUT, 128, 128 };
            wds[nw++] = { m1w  + (size_t)l * 128 * 512, base + W_M1,  128, 512 };
            wds[nw++] = { m2w  + (size_t)l * 512 * 128, base + W_M2,  512, 128 };
        }
    }
    wds[nw++] = { (const float*)d_in[5], W_GNN, 256, 128 };
    for (int i = 0; i < nw; ++i) {
        int total = wds[i].K * wds[i].N;
        transposew_kernel<<<(total + 255) / 256, 256, 0, stream>>>(
            wds[i].src, wbuf + wds[i].off, wds[i].K, wds[i].N);
    }
    fuse_enc_kernel<<<1, 128, 0, stream>>>(
        (const float*)d_in[1], (const float*)d_in[2], (const float*)d_in[3],
        (const float*)d_in[4], Af, cef, nefw);

    // 2) encoder -> feat, rm
    encoder_kernel<<<N_NODES * MAX_E, 128, 0, stream>>>(
        edge_vectors, nef_indices, nef_mask, centers, neighbors, species,
        Af, cef, nefw, feat, rm);

    // 3) transformer t0 on feat (in place)
    {
        int b = 15;
        transformer_kernel<<<N_NODES, 256, 0, stream>>>(
            feat, rm, wbuf,
            (const float*)d_in[b + 0], (const float*)d_in[b + 1],
            (const float*)d_in[b + 3], (const float*)d_in[b + 5],
            (const float*)d_in[b + 6], (const float*)d_in[b + 7],
            (const float*)d_in[b + 9], (const float*)d_in[b + 11]);
    }

    // 4) message passing into t1 input (zero masked slots first)
    hipMemsetAsync(t1in, 0, (size_t)N_NODES * MAX_E * DIM * sizeof(float), stream);
    gnn_kernel<<<N_EDGES / 64, 256, 0, stream>>>(feat, centers, n2e, corr,
                                                 wbuf + W_GNN, t1in);

    // 5) transformer t1 on t1in (in place)
    {
        int b = 27;
        transformer_kernel<<<N_NODES, 256, 0, stream>>>(
            t1in, rm, wbuf + PT,
            (const float*)d_in[b + 0], (const float*)d_in[b + 1],
            (const float*)d_in[b + 3], (const float*)d_in[b + 5],
            (const float*)d_in[b + 6], (const float*)d_in[b + 7],
            (const float*)d_in[b + 9], (const float*)d_in[b + 11]);
    }

    // 6) masked readout
    readout_kernel<<<N_NODES, 128, 0, stream>>>(feat, t1in, rm, last_w, comp_w,
                                                species, (float*)d_out);
}
